// Linear_75187697484301
// MI455X (gfx1250) — matrix operand order repaired and checked by reading, not compiled, not run on hardware
//
#include <hip/hip_runtime.h>
#include <hip/hip_bf16.h>

typedef __attribute__((ext_vector_type(16))) __bf16 v16bf;
typedef __attribute__((ext_vector_type(8)))  __bf16 bf8v;
typedef __attribute__((ext_vector_type(8)))  float  v8f;

#define IN_F  4096
#define OUT_F 4096

// ---------------------------------------------------------------------------
// CDNA5 async global->LDS (ASYNCcnt-tracked), per ISA 15.18.3 op 98.
// ---------------------------------------------------------------------------
__device__ __forceinline__ void async_load_b128(unsigned lds_off, const void* gaddr)
{
    asm volatile("global_load_async_to_lds_b128 %0, %1, off"
                 :: "v"(lds_off), "v"(gaddr) : "memory");
}
__device__ __forceinline__ void wait_async0()
{
    asm volatile("s_wait_asynccnt 0x0" ::: "memory");
}

// ---------------------------------------------------------------------------
// Cayley: Q = (I+S)^-1 (I-S), S = 0.5(R - R^T).  128 blocks of 64x64.
// ---------------------------------------------------------------------------
__global__ __launch_bounds__(64) void cayley_kernel(const float* __restrict__ R,
                                                    float* __restrict__ Q)
{
    __shared__ __align__(16) float As[64][66];
    __shared__ __align__(16) float Xs[64][66];
    const int b = blockIdx.x;                 // level*64 + d
    const float* Rb = R + (size_t)b * 64 * 64;
    const int j = threadIdx.x;

    for (int c = 0; c < 64; ++c) {
        float s  = 0.5f * (Rb[j * 64 + c] - Rb[c * 64 + j]);
        float id = (j == c) ? 1.0f : 0.0f;
        As[j][c] = id + s;
        Xs[j][c] = id - s;
    }
    __syncthreads();

    for (int k = 0; k < 64; ++k) {
        if (j != k) {
            float f = As[j][k] / As[k][k];
            for (int c = k; c < 64; ++c) As[j][c] -= f * As[k][c];
            for (int c = 0; c < 64; ++c) Xs[j][c] -= f * Xs[k][c];
        }
        __syncthreads();
    }
    float inv = 1.0f / As[j][j];
    for (int c = 0; c < 64; ++c)
        Q[(size_t)b * 4096 + j * 64 + c] = Xs[j][c] * inv;
}

// ---------------------------------------------------------------------------
// bf16x3 split helpers for fp32-staged tiles (used only in block_rot).
// ---------------------------------------------------------------------------
__device__ inline void split_fragA(const float* __restrict__ row, int lane,
                                   v16bf& hi, v16bf& lo)
{
    const int kb = ((lane >> 4) & 1) * 16;
#pragma unroll
    for (int e = 0; e < 8; ++e) {
        float f0 = row[kb + e];
        float f1 = row[kb + 8 + e];
        __bf16 h0 = (__bf16)f0, h1 = (__bf16)f1;
        hi[e]     = h0;  hi[8 + e] = h1;
        lo[e]     = (__bf16)(f0 - (float)h0);
        lo[8 + e] = (__bf16)(f1 - (float)h1);
    }
}

__device__ inline void split_fragB(const float* __restrict__ row, int lane,
                                   v16bf& hi, v16bf& lo)
{
    const int kb = ((lane >> 4) & 1) * 16;
#pragma unroll
    for (int e = 0; e < 16; ++e) {
        float f = row[kb + e];
        __bf16 h = (__bf16)f;
        hi[e] = h;
        lo[e] = (__bf16)(f - (float)h);
    }
}

// acc += Ahi*Bhi + Ahi*Blo + Alo*Bhi   (fp32-accurate bf16x3)
__device__ inline v8f wmma3(v8f acc, v16bf ahi, v16bf alo, v16bf bhi, v16bf blo)
{
    acc = __builtin_amdgcn_wmma_f32_16x16x32_bf16(false, ahi, false, bhi, (short)0, acc, false, false);
    acc = __builtin_amdgcn_wmma_f32_16x16x32_bf16(false, ahi, false, blo, (short)0, acc, false, false);
    acc = __builtin_amdgcn_wmma_f32_16x16x32_bf16(false, alo, false, bhi, (short)0, acc, false, false);
    return acc;
}

// Butterfly permutation for level 1 == swap bits 5 and 6 of the column index.
template<int LEVEL>
__device__ inline int pcol(int d, int c)
{
    int j = d * 64 + c;
    if (LEVEL == 0) return j;
    return (j & ~0x60) | ((j & 0x20) << 1) | ((j & 0x40) >> 1);
}

// ---------------------------------------------------------------------------
// Block rotation: Wout[:, P(d*64+c)] = sum_k Win[:, P(d*64+k)] * Q[d][c][k]
// LEVEL 0: fp32 -> fp32 (W1).  LEVEL 1: fp32 -> (bf16 hi, bf16 lo) * boft_s.
// Tile M=128, N=64, K=64; 8 waves of 32x32.  Safe in-place on Win (LEVEL 1
// reads of a workgroup's rows/cols all precede its writes).
// ---------------------------------------------------------------------------
template<int LEVEL>
__global__ __launch_bounds__(256) void block_rot_kernel(
    const float* __restrict__ Win, const float* __restrict__ Qall,
    const float* __restrict__ s, float* __restrict__ WoutF,
    __bf16* __restrict__ WoutHi, __bf16* __restrict__ WoutLo)
{
    __shared__ __align__(16) float Atile[128][36];
    __shared__ __align__(16) float Btile[64][36];
    const int mtile = blockIdx.x;            // 0..31 (rows of W)
    const int d     = blockIdx.y;            // 0..63 (block index)
    const int t     = threadIdx.x;
    const int lane  = t & 31;
    const int wave  = t >> 5;
    const int wm = wave & 3, wn = wave >> 2;
    const float* Qd = Qall + ((size_t)LEVEL * 64 + d) * 4096;

    v8f acc[2][2] = {};

    for (int kt = 0; kt < 64; kt += 32) {
        const int colbase = pcol<LEVEL>(d, kt);   // 32 contiguous source cols
#pragma unroll
        for (int i = 0; i < 4; ++i) {
            int idx = t + 256 * i;
            int r = idx >> 3, q = idx & 7;
            *(float4*)(&Atile[r][q * 4]) =
                *(const float4*)(Win + (size_t)(mtile * 128 + r) * IN_F + colbase + q * 4);
        }
#pragma unroll
        for (int i = 0; i < 2; ++i) {
            int idx = t + 256 * i;
            int r = idx >> 3, q = idx & 7;
            *(float4*)(&Btile[r][q * 4]) =
                *(const float4*)(Qd + (size_t)r * 64 + kt + q * 4);
        }
        __syncthreads();

        v16bf ahi[2], alo[2], bhi[2], blo[2];
#pragma unroll
        for (int mi = 0; mi < 2; ++mi)
            split_fragA(&Atile[wm * 32 + mi * 16 + (lane & 15)][0], lane, ahi[mi], alo[mi]);
#pragma unroll
        for (int nj = 0; nj < 2; ++nj)
            split_fragB(&Btile[wn * 32 + nj * 16 + (lane & 15)][0], lane, bhi[nj], blo[nj]);
#pragma unroll
        for (int mi = 0; mi < 2; ++mi)
#pragma unroll
            for (int nj = 0; nj < 2; ++nj)
                acc[mi][nj] = wmma3(acc[mi][nj], ahi[mi], alo[mi], bhi[nj], blo[nj]);
        __syncthreads();
    }

    const int r0 = ((lane >> 4) & 1) * 8;
    const int cn = lane & 15;
#pragma unroll
    for (int mi = 0; mi < 2; ++mi)
#pragma unroll
        for (int nj = 0; nj < 2; ++nj) {
            int clocal = wn * 32 + nj * 16 + cn;
            int gcol   = pcol<LEVEL>(d, clocal);
#pragma unroll
            for (int v = 0; v < 8; ++v) {
                int grow = mtile * 128 + wm * 32 + mi * 16 + r0 + v;
                float val = acc[mi][nj][v];
                size_t idx = (size_t)grow * IN_F + gcol;
                if (LEVEL == 0) {
                    WoutF[idx] = val;
                } else {
                    val *= s[grow];
                    __bf16 h = (__bf16)val;
                    WoutHi[idx] = h;
                    WoutLo[idx] = (__bf16)(val - (float)h);
                }
            }
        }
}

// ---------------------------------------------------------------------------
// Split x into bf16 hi/lo in natural k order (no swizzle); the GEMM takes k = 16h + s
// so an A-fragment is one contiguous 32B LDS read per lane.
// Each thread handles 32 consecutive floats.
// ---------------------------------------------------------------------------
__global__ __launch_bounds__(256) void split_x_kernel(const float* __restrict__ X,
                                                      __bf16* __restrict__ Hi,
                                                      __bf16* __restrict__ Lo)
{
    const size_t base = ((size_t)blockIdx.x * 256 + threadIdx.x) * 32;
    bf8v h[4], l[4];
#pragma unroll
    for (int c = 0; c < 4; ++c) {
        float4 f0 = *(const float4*)(X + base + c * 8);
        float4 f1 = *(const float4*)(X + base + c * 8 + 4);
        float fs[8] = {f0.x, f0.y, f0.z, f0.w, f1.x, f1.y, f1.z, f1.w};
#pragma unroll
        for (int e = 0; e < 8; ++e) {
            __bf16 hh = (__bf16)fs[e];
            h[c][e] = hh;
            l[c][e] = (__bf16)(fs[e] - (float)hh);
        }
    }
    // source chunk c -> dest chunk c  (natural k order)
    *(bf8v*)(Hi + base +  0) = h[0];
    *(bf8v*)(Hi + base +  8) = h[1];
    *(bf8v*)(Hi + base + 16) = h[2];
    *(bf8v*)(Hi + base + 24) = h[3];
    *(bf8v*)(Lo + base +  0) = l[0];
    *(bf8v*)(Lo + base +  8) = l[1];
    *(bf8v*)(Lo + base + 16) = l[2];
    *(bf8v*)(Lo + base + 24) = l[3];
}

// ---------------------------------------------------------------------------
// Main GEMM: Out[m][o] = sum_k X[m][k]*W[o][k] + bias[o], 4096^3, bf16x3.
// C tile 128x128, 8 waves of 64x32, K step 32.  Double-buffered LDS filled
// with global_load_async_to_lds_b128; inner loop is ds_load + v_wmma only.
// ---------------------------------------------------------------------------
__global__ __launch_bounds__(256) void gemm_kernel(
    const __bf16* __restrict__ Ahi, const __bf16* __restrict__ Alo,
    const __bf16* __restrict__ Bhi, const __bf16* __restrict__ Blo,
    const float* __restrict__ bias, float* __restrict__ Out)
{
    __shared__ __align__(16) __bf16 lds[2][4][128][40];   // 80KB, double buffer
    const int ntile = blockIdx.x;
    const int mtile = blockIdx.y;
    const int t     = threadIdx.x;
    const int lane  = t & 31;
    const int wave  = t >> 5;
    const int wm = wave & 1;                 // 2 x 64 rows
    const int wn = wave >> 1;                // 4 x 32 cols
    const int ln = lane & 15;
    const int lh = (lane >> 4) & 1;

    const __bf16* srcs[4] = { Ahi + (size_t)mtile * 128 * IN_F,
                              Alo + (size_t)mtile * 128 * IN_F,
                              Bhi + (size_t)ntile * 128 * IN_F,
                              Blo + (size_t)ntile * 128 * IN_F };

    auto stage = [&](int buf, int ko) {
#pragma unroll
        for (int a = 0; a < 4; ++a) {
#pragma unroll
            for (int i = 0; i < 2; ++i) {
                int idx = t + 256 * i;              // 0..511
                int r = idx >> 2, q = idx & 3;      // row, 8-elem chunk
                unsigned loff = (unsigned)(size_t)&lds[buf][a][r][q * 8];
                const void* g = srcs[a] + (size_t)r * IN_F + ko + q * 8;
                async_load_b128(loff, g);
            }
        }
    };

    v8f acc[4][2] = {};

    stage(0, 0);
    wait_async0();
    __syncthreads();

    for (int kt = 0; kt < 128; ++kt) {
        const int cur = kt & 1;
        if (kt + 1 < 128) stage(cur ^ 1, (kt + 1) * 32);

        v16bf ahi[4], alo[4], bhi[2], blo[2];
#pragma unroll
        for (int mi = 0; mi < 4; ++mi) {
            int r = wm * 64 + mi * 16 + ln;
            ahi[mi] = *(const v16bf*)&lds[cur][0][r][lh * 16];
            alo[mi] = *(const v16bf*)&lds[cur][1][r][lh * 16];
        }
#pragma unroll
        for (int nj = 0; nj < 2; ++nj) {
            int n = wn * 32 + nj * 16 + ln;
            bhi[nj] = *(const v16bf*)&lds[cur][2][n][lh * 16];
            blo[nj] = *(const v16bf*)&lds[cur][3][n][lh * 16];
        }
#pragma unroll
        for (int mi = 0; mi < 4; ++mi)
#pragma unroll
            for (int nj = 0; nj < 2; ++nj)
                acc[mi][nj] = wmma3(acc[mi][nj], ahi[mi], alo[mi], bhi[nj], blo[nj]);

        wait_async0();
        __syncthreads();
    }

    const int r0 = lh * 8;
#pragma unroll
    for (int mi = 0; mi < 4; ++mi)
#pragma unroll
        for (int nj = 0; nj < 2; ++nj) {
            int gcol = ntile * 128 + wn * 32 + nj * 16 + ln;
            float bv = bias[gcol];
#pragma unroll
            for (int v = 0; v < 8; ++v) {
                int grow = mtile * 128 + wm * 64 + mi * 16 + r0 + v;
                Out[(size_t)grow * OUT_F + gcol] = acc[mi][nj][v] + bv;
            }
        }
}

// ---------------------------------------------------------------------------
extern "C" void kernel_launch(void* const* d_in, const int* in_sizes, int n_in,
                              void* d_out, int out_size, void* d_ws, size_t ws_size,
                              hipStream_t stream)
{
    (void)in_sizes; (void)n_in; (void)out_size; (void)ws_size;
    const float* x      = (const float*)d_in[0];   // (2,2048,4096)
    const float* weight = (const float*)d_in[1];   // (4096,4096)
    const float* bias   = (const float*)d_in[2];   // (4096,)
    const float* boft_R = (const float*)d_in[3];   // (2,64,64,64)
    const float* boft_s = (const float*)d_in[4];   // (4096,1)
    float* out = (float*)d_out;

    // ws layout (130MB):
    //   [0, 2MB)     Q fp32 (2*64*64*64)
    //   [2, 66MB)    W1 fp32 (4096x4096); reused as Xhi|Xlo bf16 after level 1
    //   [66, 98MB)   Whi bf16
    //   [98, 130MB)  Wlo bf16
    float*  Qw  = (float*)d_ws;
    float*  W1  = (float*)((char*)d_ws + ((size_t)2 << 20));
    __bf16* Xhi = (__bf16*)W1;
    __bf16* Xlo = Xhi + (size_t)4096 * IN_F;
    __bf16* Whi = (__bf16*)((char*)d_ws + ((size_t)66 << 20));
    __bf16* Wlo = (__bf16*)((char*)d_ws + ((size_t)98 << 20));

    cayley_kernel<<<dim3(128), dim3(64), 0, stream>>>(boft_R, Qw);
    block_rot_kernel<0><<<dim3(32, 64), dim3(256), 0, stream>>>(
        weight, Qw, boft_s, W1, nullptr, nullptr);
    block_rot_kernel<1><<<dim3(32, 64), dim3(256), 0, stream>>>(
        W1, Qw, boft_s, nullptr, Whi, Wlo);
    split_x_kernel<<<dim3(2048), dim3(256), 0, stream>>>(x, Xhi, Xlo);
    gemm_kernel<<<dim3(32, 32), dim3(256), 0, stream>>>(Xhi, Xlo, Whi, Wlo, bias, out);
}
